// GAT_T_49014166782488
// MI455X (gfx1250) — matrix operand order repaired and checked by reading, not compiled, not run on hardware
//
#include <hip/hip_runtime.h>
#include <hip/hip_bf16.h>
#include <math.h>

typedef __bf16 bf16;
typedef __attribute__((ext_vector_type(16))) __bf16 v16bf;
typedef __attribute__((ext_vector_type(8)))  float  v8f;

union FragU { v16bf v; unsigned u[8]; };

// ---------------------------------------------------------------------------
// WMMA helpers (gfx1250 wave32, V_WMMA_F32_16X16X32_BF16)
// ---------------------------------------------------------------------------
__device__ inline v16bf load_frag_a(const bf16* lds, int row0, int ld) {
  int lane = threadIdx.x & 31;
  int m    = row0 + (lane & 15);
  int half = lane >> 4;
  const unsigned* p = (const unsigned*)(lds + (size_t)m * ld);
  FragU f;
#pragma unroll
  for (int r = 0; r < 8; ++r) f.u[r] = p[r + 8 * half];           // K = 2r + 16h
  return f.v;
}

__device__ inline v16bf load_frag_b(const bf16* lds, int n0, int ld) {
  int lane = threadIdx.x & 31;
  int n    = n0 + (lane & 15);
  int half = lane >> 4;
  const unsigned* p = (const unsigned*)(lds + (size_t)n * ld);
  FragU f;
#pragma unroll
  for (int r = 0; r < 8; ++r) f.u[r] = p[r + 8 * half];           // K = 2r + 16h
  return f.v;
}

__device__ inline v8f wmma_bf16(v16bf a, v16bf b, v8f c) {
  return __builtin_amdgcn_wmma_f32_16x16x32_bf16(false, a, false, b,
                                                 (short)0, c, false, false);
}

// CDNA5 async global->LDS copy (ASYNCcnt-tracked), 16B per lane
__device__ inline void async_cp16(void* lds_ptr, const void* gptr) {
  unsigned l = (unsigned)(size_t)lds_ptr;   // LDS byte address = addr[31:0]
  asm volatile("global_load_async_to_lds_b128 %0, %1, off"
               :: "v"(l), "v"(gptr) : "memory");
}
__device__ inline void async_wait0() {
  asm volatile("s_wait_asynccnt 0x0" ::: "memory");
}

// order-preserving float <-> uint for atomicMax-based segment max
__device__ inline unsigned f2ord(float f) {
  unsigned b = __float_as_uint(f);
  return (b & 0x80000000u) ? ~b : (b | 0x80000000u);
}
__device__ inline float ord2f(unsigned u) {
  return __uint_as_float((u & 0x80000000u) ? (u & 0x7fffffffu) : ~u);
}
#define ORD_NEG_INF 0x007FFFFFu   // f2ord(-inf)

// ---------------------------------------------------------------------------
// bf16 GEMM: C[M,N] = A[M,K] @ Wsel[N,K]^T + bias   (A, W bf16; C, bias f32)
// W/bias selected by output row tile (< splitW -> W1/b1 else W2/b2).
// 256 thr = 8 waves, tile 64x64, K chunks of 64, double-buffered async
// global->LDS staging, 4 WMMAs / wave / chunk, 1 barrier / chunk.
// ---------------------------------------------------------------------------
#define GT_M 64
#define GT_N 64
#define GT_K 64

__global__ __launch_bounds__(256)
void gemm_bf16_kernel(const bf16* __restrict__ A,
                      const bf16* __restrict__ W1, const bf16* __restrict__ W2,
                      const float* __restrict__ b1, const float* __restrict__ b2,
                      int splitW,
                      float* __restrict__ C, int M, int N, int K)
{
  __shared__ __align__(16) bf16 sA[2][GT_M][GT_K];   // 2 x 8 KB
  __shared__ __align__(16) bf16 sB[2][GT_N][GT_K];   // 2 x 8 KB

  int tid  = threadIdx.x;
  int wave = tid >> 5;
  int mi   = wave & 3;            // 4 M subtiles
  int nj0  = (wave >> 2) * 2;     // 2 of 4 N subtiles per wave
  int tileM = blockIdx.x * GT_M;
  int tileN = blockIdx.y * GT_N;

  const bf16*  W    = (tileM < splitW) ? W1 : W2;
  const float* bias = (tileM < splitW) ? b1 : b2;

  auto stage = [&](int b, int k0) {
    // A tile: 64 rows x 128B ; B tile likewise. 512 x 16B copies each, 2/thread.
    for (int i = tid; i < 512; i += 256) {
      int r = i >> 3, c = (i & 7) << 3;
      async_cp16(&sA[b][r][c], A + (size_t)(tileM + r) * K + k0 + c);
    }
    for (int i = tid; i < 512; i += 256) {
      int r = i >> 3, c = (i & 7) << 3;
      async_cp16(&sB[b][r][c], W + (size_t)(tileN + r) * K + k0 + c);
    }
  };

  v8f acc[2] = {};
  stage(0, 0);
  async_wait0();
  __syncthreads();

  int cur = 0;
  for (int k0 = 0; k0 < K; k0 += GT_K) {
    if (k0 + GT_K < K) stage(cur ^ 1, k0 + GT_K);     // prefetch next chunk
#pragma unroll
    for (int ks = 0; ks < GT_K; ks += 32) {
      v16bf fa  = load_frag_a(&sA[cur][0][0] + ks, mi * 16, GT_K);
      v16bf fb0 = load_frag_b(&sB[cur][0][0] + ks, (nj0 + 0) * 16, GT_K);
      v16bf fb1 = load_frag_b(&sB[cur][0][0] + ks, (nj0 + 1) * 16, GT_K);
      acc[0] = wmma_bf16(fa, fb0, acc[0]);
      acc[1] = wmma_bf16(fa, fb1, acc[1]);
    }
    async_wait0();
    __syncthreads();
    cur ^= 1;
  }

  int lane = tid & 31, half = lane >> 4;
#pragma unroll
  for (int j2 = 0; j2 < 2; ++j2) {
    int col = tileN + (nj0 + j2) * 16 + (lane & 15);
    float bv = bias ? bias[col] : 0.f;
#pragma unroll
    for (int j = 0; j < 8; ++j) {
      int row = tileM + mi * 16 + j + 8 * half;
      C[(size_t)row * N + col] = acc[j2][j] + bv;
    }
  }
}

// ---------------------------------------------------------------------------
// per-row dots: hs[r]=h[r,:].a_src, hd[r]=h[r,:].a_dst ; one wave per row
// ---------------------------------------------------------------------------
__global__ __launch_bounds__(256)
void rowdot_kernel(const float* __restrict__ h, const float* __restrict__ a_src,
                   const float* __restrict__ a_dst, float* hs, float* hd,
                   int n, int H)
{
  int wave = threadIdx.x >> 5;
  int lane = threadIdx.x & 31;
  int row  = blockIdx.x * 8 + wave;
  if (row >= n) return;
  const float* hp = h + (size_t)row * H;
  float s = 0.f, d = 0.f;
  for (int c = lane; c < H; c += 32) {
    float v = hp[c];
    s += v * a_src[c];
    d += v * a_dst[c];
  }
#pragma unroll
  for (int off = 16; off > 0; off >>= 1) {
    s += __shfl_down(s, off, 32);
    d += __shfl_down(d, off, 32);
  }
  if (lane == 0) { hs[row] = s; hd[row] = d; }
}

// homogeneous edge decode: [edge_AB ; edge_BA (src+NA) ; self-loops]
__device__ inline void edge_decode(int t, const int* __restrict__ eAB,
                                   const int* __restrict__ eBA,
                                   int E, int na, int& s, int& d)
{
  if (t < E)          { s = eAB[t];          d = eAB[E + t] + na; }
  else if (t < 2 * E) { s = eBA[t - E] + na; d = eBA[t];          } // eBA[E + (t-E)]
  else                { s = t - 2 * E;       d = s;               }
}

__global__ void edge_max_kernel(const int* __restrict__ eAB, const int* __restrict__ eBA,
                                const float* __restrict__ hs, const float* __restrict__ hd,
                                unsigned* __restrict__ mo, float* __restrict__ eraw,
                                int E, int na, int EE)
{
  int t = blockIdx.x * blockDim.x + threadIdx.x;
  if (t >= EE) return;
  int s, d; edge_decode(t, eAB, eBA, E, na, s, d);
  float e = hs[s] + hd[d];
  e = (e > 0.f) ? e : 0.2f * e;                 // LeakyReLU(0.2)
  eraw[t] = e;
  atomicMax(mo + d, f2ord(e));
}

__global__ void edge_expsum_kernel(const int* __restrict__ eAB, const int* __restrict__ eBA,
                                   const unsigned* __restrict__ mo, float* __restrict__ denom,
                                   float* __restrict__ eraw, int E, int na, int EE)
{
  int t = blockIdx.x * blockDim.x + threadIdx.x;
  if (t >= EE) return;
  int s, d; edge_decode(t, eAB, eBA, E, na, s, d);
  float ex = __expf(eraw[t] - ord2f(mo[d]));
  eraw[t] = ex;
  atomicAdd(denom + d, ex);
}

__global__ void edge_scatter_kernel(const int* __restrict__ eAB, const int* __restrict__ eBA,
                                    const float* __restrict__ h, const float* __restrict__ eraw,
                                    const float* __restrict__ denom, float* __restrict__ out,
                                    int E, int na, int EE, int H)
{
  int idx = blockIdx.x * blockDim.x + threadIdx.x;
  int t   = idx >> 7;                           // H/4 = 128 chunks per edge
  if (t >= EE) return;
  int fc  = (idx & 127) << 2;
  int s, d; edge_decode(t, eAB, eBA, E, na, s, d);
  float alpha = eraw[t] / denom[d];
  const float4 hv = *(const float4*)(h + (size_t)s * H + fc);
  float* op = out + (size_t)d * H + fc;
  atomicAdd(op + 0, hv.x * alpha);
  atomicAdd(op + 1, hv.y * alpha);
  atomicAdd(op + 2, hv.z * alpha);
  atomicAdd(op + 3, hv.w * alpha);
}

// utility kernels
__global__ void fill_f32_kernel(float* p, float v, int count) {
  int i = blockIdx.x * blockDim.x + threadIdx.x;
  if (i < count) p[i] = v;
}
__global__ void fill_u32_kernel(unsigned* p, unsigned v, int count) {
  int i = blockIdx.x * blockDim.x + threadIdx.x;
  if (i < count) p[i] = v;
}
__global__ void init_out_bias_kernel(float* out, const float* __restrict__ bias,
                                     int n, int H) {
  int i = blockIdx.x * blockDim.x + threadIdx.x;
  if (i < n * H) out[i] = bias[i & (H - 1)];
}
__global__ void concat_kernel(const float* __restrict__ l, const float* __restrict__ g,
                              float* __restrict__ out, int n, int H) {
  int i = blockIdx.x * blockDim.x + threadIdx.x;
  if (i >= n * H) return;
  int r = i / H, c = i - r * H;
  out[(size_t)r * (2 * H) + c]     = l[i];
  out[(size_t)r * (2 * H) + H + c] = g[i];
}
__global__ void f32_to_bf16_kernel(const float* __restrict__ in, bf16* __restrict__ o,
                                   int count) {
  int i = blockIdx.x * blockDim.x + threadIdx.x;
  if (i < count) o[i] = (bf16)in[i];
}

// ---------------------------------------------------------------------------
// Flash attention: out = softmax(Q K^T / sqrt(512)) V, single head, d=512.
// qkvh: bf16 [n][1536] (q|k|v).  256 thr = 8 waves, 32 queries/block,
// 32-key chunks.  Waves 0-3 each compute one 16x16 S subtile (16 WMMAs over
// d=512); every wave owns 64 output columns: 8 P.V WMMAs / chunk.
// Q/K staged via async global->LDS b128 copies; V transposed via LDS stores.
// Dynamic LDS ~102 KB (3 blocks/WGP of the 320 KB pool).
// ---------------------------------------------------------------------------
#define FA_M  32
#define FA_NK 32
#define FA_D  512
#define FA_SMEM (FA_M*FA_D*2 + FA_NK*FA_D*2 + FA_D*FA_NK*2 + FA_M*FA_NK*4 + FA_M*FA_NK*2 + FA_M*3*4)

__global__ __launch_bounds__(256)
void flash_attn_kernel(const bf16* __restrict__ qkvh, float* __restrict__ out, int n)
{
  extern __shared__ __align__(16) char smem[];
  bf16*  sQ   = (bf16*)smem;                    // [32][512]  32 KB
  bf16*  sK   = sQ  + FA_M * FA_D;              // [32][512]  32 KB
  bf16*  sVt  = sK  + FA_NK * FA_D;             // [512][32]  32 KB
  float* sS   = (float*)(sVt + FA_D * FA_NK);   // [32][32]    4 KB
  bf16*  sP   = (bf16*)(sS + FA_M * FA_NK);     // [32][32]    2 KB
  float* sMax = (float*)(sP + FA_M * FA_NK);
  float* sSum = sMax + FA_M;
  float* sCorr= sSum + FA_M;

  int tid  = threadIdx.x;
  int wave = tid >> 5, lane = tid & 31, half = lane >> 4;
  int row0 = blockIdx.x * FA_M;
  int wcol = wave * 64;
  const float scale = 0.044194173824159216f;    // 1/sqrt(512)

  // async-stage Q tile (16B per copy; 8 copies per thread)
  for (int i = tid; i < (FA_M * FA_D) / 8; i += 256) {
    int r = i >> 6, c = (i & 63) << 3;
    async_cp16(sQ + r * FA_D + c, qkvh + (size_t)(row0 + r) * 1536 + c);
  }
  if (tid < FA_M) { sMax[tid] = -1e30f; sSum[tid] = 0.f; }
  async_wait0();
  __syncthreads();

  v8f acc[2][4] = {};
  for (int kb = 0; kb < n; kb += FA_NK) {
    // stage K [key][d] async; stage V transposed [d][key] via LDS stores
    for (int i = tid; i < (FA_NK * FA_D) / 8; i += 256) {
      int r = i >> 6, c = (i & 63) << 3;
      async_cp16(sK + r * FA_D + c, qkvh + (size_t)(kb + r) * 1536 + 512 + c);
      uint4 pv = *(const uint4*)(qkvh + (size_t)(kb + r) * 1536 + 1024 + c);
      const bf16* pb = (const bf16*)&pv;
#pragma unroll
      for (int e = 0; e < 8; ++e) sVt[(c + e) * FA_NK + r] = pb[e];
    }
    async_wait0();
    __syncthreads();
    // S = scale * Q K^T  (waves 0-3, one 16x16 subtile each)
    if (wave < 4) {
      int smi = wave & 1, sni = wave >> 1;
      v8f sacc = {};
#pragma unroll
      for (int kk = 0; kk < FA_D; kk += 32) {
        v16bf fa = load_frag_a(sQ + kk, smi * 16, FA_D);
        v16bf fb = load_frag_b(sK + kk, sni * 16, FA_D);
        sacc = wmma_bf16(fa, fb, sacc);
      }
      int cc = sni * 16 + (lane & 15);
#pragma unroll
      for (int j = 0; j < 8; ++j)
        sS[(smi * 16 + j + 8 * half) * FA_NK + cc] = sacc[j] * scale;
    }
    __syncthreads();
    // online softmax, one thread per query row
    if (tid < FA_M) {
      int r = tid;
      float mo = sMax[r], mn = mo;
#pragma unroll
      for (int j = 0; j < FA_NK; ++j) mn = fmaxf(mn, sS[r * FA_NK + j]);
      float corr = __expf(mo - mn);
      float ssum = 0.f;
#pragma unroll
      for (int j = 0; j < FA_NK; ++j) {
        float p = __expf(sS[r * FA_NK + j] - mn);
        sP[r * FA_NK + j] = (bf16)p;
        ssum += p;
      }
      sMax[r] = mn; sSum[r] = sSum[r] * corr + ssum; sCorr[r] = corr;
    }
    __syncthreads();
    // rescale accumulators, O += P V
#pragma unroll
    for (int smi = 0; smi < 2; ++smi) {
#pragma unroll
      for (int ni = 0; ni < 4; ++ni)
#pragma unroll
        for (int j = 0; j < 8; ++j)
          acc[smi][ni][j] *= sCorr[smi * 16 + j + 8 * half];
      v16bf fp = load_frag_a(sP, smi * 16, FA_NK);
#pragma unroll
      for (int ni = 0; ni < 4; ++ni) {
        v16bf fv = load_frag_b(sVt, wcol + ni * 16, FA_NK);
        acc[smi][ni] = wmma_bf16(fp, fv, acc[smi][ni]);
      }
    }
    __syncthreads();   // sK/sVt/sP reused next chunk
  }
#pragma unroll
  for (int smi = 0; smi < 2; ++smi)
#pragma unroll
    for (int ni = 0; ni < 4; ++ni) {
      int col = wcol + ni * 16 + (lane & 15);
#pragma unroll
      for (int j = 0; j < 8; ++j) {
        int r = smi * 16 + j + 8 * half;
        out[(size_t)(row0 + r) * FA_D + col] = acc[smi][ni][j] / sSum[r];
      }
    }
}

// ---------------------------------------------------------------------------
extern "C" void kernel_launch(void* const* d_in, const int* in_sizes, int n_in,
                              void* d_out, int out_size, void* d_ws, size_t ws_size,
                              hipStream_t stream)
{
  (void)in_sizes; (void)n_in; (void)out_size; (void)ws_size;

  const int NA = 4096, N = 8192, IN = 256, H = 512, E = 131072;
  const int EE = 2 * E + N;

  const float* x_A    = (const float*)d_in[0];
  const float* x_B    = (const float*)d_in[1];
  const int*   eAB    = (const int*)  d_in[2];
  const int*   eBA    = (const int*)  d_in[3];
  const float* W_inA  = (const float*)d_in[4];
  const float* b_inA  = (const float*)d_in[5];
  const float* W_inB  = (const float*)d_in[6];
  const float* b_inB  = (const float*)d_in[7];
  const float* W_in2A = (const float*)d_in[8];
  const float* b_in2A = (const float*)d_in[9];
  const float* W_in2B = (const float*)d_in[10];
  const float* b_in2B = (const float*)d_in[11];
  const float* Wg[2]    = {(const float*)d_in[12], (const float*)d_in[16]};
  const float* a_src[2] = {(const float*)d_in[13], (const float*)d_in[17]};
  const float* a_dst[2] = {(const float*)d_in[14], (const float*)d_in[18]};
  const float* bg[2]    = {(const float*)d_in[15], (const float*)d_in[19]};
  const float* Wqkv   = (const float*)d_in[20];
  const float* bqkv   = (const float*)d_in[21];
  const float* Wo     = (const float*)d_in[22];
  const float* bo     = (const float*)d_in[23];

  float* ws = (float*)d_ws;
  float* buf_l   = ws; ws += (size_t)N * H;
  float* buf_l2  = ws; ws += (size_t)N * H;
  float* buf_h   = ws; ws += (size_t)N * H;
  float* buf_g   = ws; ws += (size_t)N * H;
  float* buf_qkv = ws; ws += (size_t)N * 3 * H;
  float* buf_t   = ws; ws += (size_t)N * H;
  float* hs      = ws; ws += N;
  float* hd      = ws; ws += N;
  float* denom   = ws; ws += N;
  unsigned* mo   = (unsigned*)ws; ws += N;
  float* eraw    = ws; ws += EE;
  // bf16 staging areas (sizes in float units; all 16B aligned)
  bf16* qkvh  = (bf16*)ws; ws += (size_t)N * 3 * H / 2;   // qkv, bf16
  bf16* xh    = (bf16*)ws; ws += (size_t)N * IN / 2;      // concat(x_A,x_B)
  bf16* acth  = (bf16*)ws; ws += (size_t)N * H / 2;       // current activation
  bf16* whInA = (bf16*)ws; ws += (size_t)H * IN / 2;
  bf16* whInB = (bf16*)ws; ws += (size_t)H * IN / 2;
  bf16* wh2A  = (bf16*)ws; ws += (size_t)H * IN / 2;
  bf16* wh2B  = (bf16*)ws; ws += (size_t)H * IN / 2;
  bf16* whG0  = (bf16*)ws; ws += (size_t)H * H / 2;
  bf16* whG1  = (bf16*)ws; ws += (size_t)H * H / 2;
  bf16* whQkv = (bf16*)ws; ws += (size_t)3 * H * H / 2;
  bf16* whO   = (bf16*)ws; ws += (size_t)H * H / 2;
  bf16* whG[2] = {whG0, whG1};

  dim3 blk(256);
  dim3 gNH(N / GT_M, H / GT_N);
  #define CVT(src, dst, cnt) \
    f32_to_bf16_kernel<<<((cnt) + 255) / 256, blk, 0, stream>>>((src), (dst), (cnt))

  // one-time bf16 conversions: inputs (concat) + all weights
  CVT(x_A, xh, NA * IN);
  CVT(x_B, xh + (size_t)NA * IN, (N - NA) * IN);
  CVT(W_inA, whInA, H * IN);   CVT(W_inB, whInB, H * IN);
  CVT(W_in2A, wh2A, H * IN);   CVT(W_in2B, wh2B, H * IN);
  CVT(Wg[0], whG0, H * H);     CVT(Wg[1], whG1, H * H);
  CVT(Wqkv, whQkv, 3 * H * H); CVT(Wo, whO, H * H);

  // input linears for both branches (weights type-split at row NA)
  gemm_bf16_kernel<<<gNH, blk, 0, stream>>>(xh, whInA, whInB, b_inA, b_inB, NA,
                                            buf_l, N, H, IN);
  gemm_bf16_kernel<<<gNH, blk, 0, stream>>>(xh, wh2A, wh2B, b_in2A, b_in2B, NA,
                                            buf_g, N, H, IN);

  // two GAT layers (local branch)
  float* lcur = buf_l;
  float* lout = buf_l2;
  for (int L = 0; L < 2; ++L) {
    CVT(lcur, acth, N * H);
    gemm_bf16_kernel<<<gNH, blk, 0, stream>>>(acth, whG[L], whG[L],
                                              nullptr, nullptr, N, buf_h, N, H, H);
    rowdot_kernel<<<N / 8, blk, 0, stream>>>(buf_h, a_src[L], a_dst[L], hs, hd, N, H);
    fill_u32_kernel<<<(N + 255) / 256, blk, 0, stream>>>(mo, ORD_NEG_INF, N);
    fill_f32_kernel<<<(N + 255) / 256, blk, 0, stream>>>(denom, 0.f, N);
    init_out_bias_kernel<<<(N * H + 255) / 256, blk, 0, stream>>>(lout, bg[L], N, H);
    edge_max_kernel<<<(EE + 255) / 256, blk, 0, stream>>>(eAB, eBA, hs, hd, mo, eraw, E, NA, EE);
    edge_expsum_kernel<<<(EE + 255) / 256, blk, 0, stream>>>(eAB, eBA, mo, denom, eraw, E, NA, EE);
    edge_scatter_kernel<<<(EE * (H / 4) + 255) / 256, blk, 0, stream>>>(
        eAB, eBA, buf_h, eraw, denom, lout, E, NA, EE, H);
    float* tmp = lcur; lcur = lout; lout = tmp;
  }
  // lcur now holds final local features

  // global branch: QKV projection, flash attention (bf16), output projection
  CVT(buf_g, acth, N * H);
  gemm_bf16_kernel<<<dim3(N / GT_M, (3 * H) / GT_N), blk, 0, stream>>>(
      acth, whQkv, whQkv, bqkv, bqkv, N, buf_qkv, N, 3 * H, H);
  CVT(buf_qkv, qkvh, N * 3 * H);
  flash_attn_kernel<<<N / FA_M, blk, FA_SMEM, stream>>>(qkvh, buf_t, N);
  CVT(buf_t, acth, N * H);
  gemm_bf16_kernel<<<gNH, blk, 0, stream>>>(acth, whO, whO, bo, bo, N,
                                            buf_h, N, H, H);

  // z = [local | global], z_A rows then z_B rows (contiguous node-major)
  concat_kernel<<<(N * H + 255) / 256, blk, 0, stream>>>(lcur, buf_h, (float*)d_out, N, H);
}
